// LSTM_24970939859003
// MI455X (gfx1250) — matrix operand order repaired and checked by reading, not compiled, not run on hardware
//
#include <hip/hip_runtime.h>
#include <stdint.h>

// ---------------------------------------------------------------------------
// LSTM on MI455X (gfx1250): persistent-kernel recurrence with bf16 WMMA.
//   - 32 workgroups x 128 threads (4 wave32), one WG per 16 hidden columns
//   - weight slice parked in LDS (per-lane blocks, DS-immediate addressing)
//   - X pre-converted to bf16 in ws when ws_size permits (removes 32x
//     duplicated f32->bf16 pack work from the sequential hot loop)
//   - c-state in registers; h via 64KB bf16 global buffer; one device-wide
//     sense-reversal barrier per timestep
// ---------------------------------------------------------------------------

typedef __attribute__((ext_vector_type(16))) __bf16 v16bf;
typedef __attribute__((ext_vector_type(8)))  float  v8f;

#define S_LEN   2048
#define BATCH   64
#define HID     512
#define IN_DIM  512
#define G4      (4 * HID)        // 2048 gate rows
#define BH      (BATCH * HID)    // 32768
#define NWG     32
#define COLS_PER_WG 16
#define THREADS 128

// ---- LDS layout: 16 per-lane blocks ---------------------------------------
// Lane ln owns rows j=0..7 (j<4: Wih gate j; j>=4: Whh gate j-4) for column
// (wg*16 + ln). Row = 520 bf16 (1040 B); block = 8 rows + 16 B pad = 8336 B.
// Lane stride 8336 B = 2084 dw == 36 (mod 64), gcd(36,64)=4 -> 16 low lanes
// of a B-fragment ds_load_b128 hit 16 distinct banks.
#define ROW_E      520                      // elements per row (512 + pad)
#define BLOCK_E    (8 * ROW_E + 8)          // 4168 elements = 8336 B
#define SMEM_BYTES (16 * BLOCK_E * 2)       // 133,376 B

// ---- fast scalar helpers --------------------------------------------------
__device__ __forceinline__ unsigned f2bf1(float f) {
    unsigned u = __float_as_uint(f);
    unsigned r = ((u >> 16) & 1u) + 0x7FFFu;
    return (u + r) >> 16;
}
__device__ __forceinline__ unsigned pkbf(float lo, float hi) {
    return (f2bf1(hi) << 16) | (f2bf1(lo) & 0xFFFFu);
}

#if __has_builtin(__builtin_amdgcn_rcpf)
__device__ __forceinline__ float fast_rcp(float x) { return __builtin_amdgcn_rcpf(x); }
#else
__device__ __forceinline__ float fast_rcp(float x) { return 1.0f / x; }
#endif

__device__ __forceinline__ float sigmoidf_(float x) {
    return fast_rcp(1.0f + __expf(-x));
}
__device__ __forceinline__ float tanhf_(float x) {
    return 1.0f - 2.0f * fast_rcp(1.0f + __expf(2.0f * x));
}

// 16 bf16 elements of a WMMA fragment, viewable as 8 dwords.
union FragB {
    v16bf    v;
    unsigned u[8];
};

__device__ __forceinline__ v8f wmma_bf16(const FragB& a, const FragB& b, v8f c) {
    return __builtin_amdgcn_wmma_f32_16x16x32_bf16(
        false, a.v, false, b.v, (short)0, c, false, false);
}

// bf16 p[0..7] and p[16..23] -> fragment (two b128 loads; works for LDS and global).
__device__ __forceinline__ FragB load_frag16(const unsigned short* p) {
    FragB f;
    uint4 t0 = *reinterpret_cast<const uint4*>(p);
    uint4 t1 = *reinterpret_cast<const uint4*>(p + 16);
    f.u[0] = t0.x; f.u[1] = t0.y; f.u[2] = t0.z; f.u[3] = t0.w;
    f.u[4] = t1.x; f.u[5] = t1.y; f.u[6] = t1.z; f.u[7] = t1.w;
    return f;
}

// ---- prep kernels ---------------------------------------------------------
// Vectorized f32 -> bf16 (RNE), 8 elements per thread. n8 = n/8.
__global__ void cvt_f32_bf16_v8(const float* __restrict__ src,
                                unsigned short* __restrict__ dst, int n8) {
    int i = blockIdx.x * blockDim.x + threadIdx.x;
    if (i < n8) {
        const float4* s = reinterpret_cast<const float4*>(src) + (size_t)i * 2;
        float4 a = s[0], b = s[1];
        uint4 o;
        o.x = pkbf(a.x, a.y); o.y = pkbf(a.z, a.w);
        o.z = pkbf(b.x, b.y); o.w = pkbf(b.z, b.w);
        reinterpret_cast<uint4*>(dst)[i] = o;
    }
}

__global__ void add_bias(const float* __restrict__ a, const float* __restrict__ b,
                         float* __restrict__ o, int n) {
    int i = blockIdx.x * blockDim.x + threadIdx.x;
    if (i < n) o[i] = a[i] + b[i];
}

// ---- persistent recurrence kernel -----------------------------------------
template <bool XBF16>
__global__ void __launch_bounds__(THREADS, 1)
lstm_persistent(const float* __restrict__ x32,            // (S,B,I) f32
                const unsigned short* __restrict__ x16,   // (S,B,I) bf16 (opt)
                const float* __restrict__ c_prev,         // (B,H) f32
                const unsigned short* __restrict__ Wih,   // (4H,I) bf16
                const unsigned short* __restrict__ Whh,   // (4H,H) bf16
                const float* __restrict__ bias,           // (4H) f32 (ih+hh)
                unsigned short* __restrict__ hbuf,        // (B,H) bf16 state
                float* __restrict__ out,                  // S*BH | h_f | c_f
                unsigned* __restrict__ sync)              // [0]=cnt, [1]=gen
{
    extern __shared__ unsigned short smem[];

    const int wave = threadIdx.x >> 5;  // 4 waves = 4 batch tiles of 16
    const int lane = threadIdx.x & 31;
    const int m0   = wave * 16;
    const int ln   = lane & 15;
    const int hi   = lane >> 4;                          // half-wave select
    const int n    = blockIdx.x * COLS_PER_WG + ln;      // hidden column
    const int b_a  = m0 + ln;                            // A-frag batch row

    // ---- one-time: stage weight slice into per-lane LDS blocks ------------
    {
        const int wgbase = blockIdx.x * COLS_PER_WG;
        for (int v = threadIdx.x; v < (128 * IN_DIM) / 8; v += THREADS) {
            const int rowL = v >> 6;            // 64 uint4-vectors per row
            const int kv   = (v & 63) * 8;
            const int m    = rowL >> 6;         // 0 = Wih, 1 = Whh
            const int g    = (rowL >> 4) & 3;
            const int nl   = rowL & 15;
            const unsigned short* src =
                (m ? Whh : Wih) + (size_t)(g * HID + wgbase + nl) * IN_DIM + kv;
            *reinterpret_cast<uint4*>(
                smem + (size_t)nl * BLOCK_E + (m * 4 + g) * ROW_E + kv) =
                *reinterpret_cast<const uint4*>(src);
        }
    }
    __syncthreads();

    // All 16 B-fragment loads come off this single base with imm offsets.
    const unsigned short* wbase = smem + (size_t)ln * BLOCK_E;

    const float bi = bias[0 * HID + n];
    const float bf = bias[1 * HID + n];
    const float bg = bias[2 * HID + n];
    const float bo = bias[3 * HID + n];

    // Persistent cell state: acc element r <-> (batch m0 + r + hi*8, col n)
    float creg[8];
#pragma unroll
    for (int r = 0; r < 8; ++r)
        creg[r] = c_prev[(size_t)(m0 + r + hi * 8) * HID + n];

    unsigned* cnt = sync;
    unsigned* gen = sync + 1;

    for (int t = 0; t < S_LEN; ++t) {
        v8f ai, af, ag, ao;
#pragma unroll
        for (int r = 0; r < 8; ++r) { ai[r] = bi; af[r] = bf; ag[r] = bg; ao[r] = bo; }

        const float*          xrow32 = x32 + (size_t)t * BH + (size_t)b_a * IN_DIM;
        const unsigned short* xrow16 = x16 + (size_t)t * BH + (size_t)b_a * IN_DIM;
        const unsigned short* hrow   = hbuf + (size_t)b_a * HID;

        if (t + 1 < S_LEN) {                       // prefetch next timestep
            if (XBF16) __builtin_prefetch(xrow16 + BH, 0, 1);
            else       __builtin_prefetch(xrow32 + BH, 0, 1);
        }

        for (int k0 = 0; k0 < IN_DIM; k0 += 32) {
            const int ca = k0 + hi * 8;    // A-frag chunk 1 (chunk 2 at +16)
            const int kb = k0 + hi * 8;    // B-frag chunk 1 (chunk 2 at +16)

            // A fragment from x
            FragB ax;
            if (XBF16) {
                ax = load_frag16(xrow16 + ca);     // overlap +16 handled below
                uint4 t1 = *reinterpret_cast<const uint4*>(xrow16 + ca + 16);
                ax.u[4] = t1.x; ax.u[5] = t1.y; ax.u[6] = t1.z; ax.u[7] = t1.w;
            } else {
                float4 f0 = *reinterpret_cast<const float4*>(xrow32 + ca);
                float4 f1 = *reinterpret_cast<const float4*>(xrow32 + ca + 4);
                float4 f2 = *reinterpret_cast<const float4*>(xrow32 + ca + 16);
                float4 f3 = *reinterpret_cast<const float4*>(xrow32 + ca + 20);
                ax.u[0] = pkbf(f0.x, f0.y); ax.u[1] = pkbf(f0.z, f0.w);
                ax.u[2] = pkbf(f1.x, f1.y); ax.u[3] = pkbf(f1.z, f1.w);
                ax.u[4] = pkbf(f2.x, f2.y); ax.u[5] = pkbf(f2.z, f2.w);
                ax.u[6] = pkbf(f3.x, f3.y); ax.u[7] = pkbf(f3.z, f3.w);
            }
            // A fragment from h (bf16)
            FragB ah;
            {
                uint4 t0 = *reinterpret_cast<const uint4*>(hrow + ca);
                uint4 t1 = *reinterpret_cast<const uint4*>(hrow + ca + 16);
                ah.u[0] = t0.x; ah.u[1] = t0.y; ah.u[2] = t0.z; ah.u[3] = t0.w;
                ah.u[4] = t1.x; ah.u[5] = t1.y; ah.u[6] = t1.z; ah.u[7] = t1.w;
            }

            ai = wmma_bf16(ax, load_frag16(wbase + 0 * ROW_E + kb), ai);
            af = wmma_bf16(ax, load_frag16(wbase + 1 * ROW_E + kb), af);
            ag = wmma_bf16(ax, load_frag16(wbase + 2 * ROW_E + kb), ag);
            ao = wmma_bf16(ax, load_frag16(wbase + 3 * ROW_E + kb), ao);
            ai = wmma_bf16(ah, load_frag16(wbase + 4 * ROW_E + kb), ai);
            af = wmma_bf16(ah, load_frag16(wbase + 5 * ROW_E + kb), af);
            ag = wmma_bf16(ah, load_frag16(wbase + 6 * ROW_E + kb), ag);
            ao = wmma_bf16(ah, load_frag16(wbase + 7 * ROW_E + kb), ao);
        }

        // Gate nonlinearities + state update (C/D layout: m = r + hi*8, n = ln)
#pragma unroll
        for (int r = 0; r < 8; ++r) {
            float iv = sigmoidf_(ai[r]);
            float fv = sigmoidf_(af[r]);
            float gv = tanhf_(ag[r]);
            float ov = sigmoidf_(ao[r]);
            float c  = fv * creg[r] + iv * gv;
            creg[r]  = c;
            float h  = ov * tanhf_(c);
            const int b = m0 + r + hi * 8;
            out[(size_t)t * BH + (size_t)b * HID + n] = h;
            hbuf[(size_t)b * HID + n] = (unsigned short)f2bf1(h);
            if (t == S_LEN - 1)
                out[(size_t)S_LEN * BH + (size_t)b * HID + n] = h;   // h_f
        }

        // Device-wide sense-reversal barrier (replay-safe: cnt returns to 0)
        __threadfence();
        __syncthreads();
        if (threadIdx.x == 0) {
            unsigned g   = __hip_atomic_load(gen, __ATOMIC_RELAXED, __HIP_MEMORY_SCOPE_AGENT);
            unsigned old = __hip_atomic_fetch_add(cnt, 1u, __ATOMIC_ACQ_REL, __HIP_MEMORY_SCOPE_AGENT);
            if (old == (unsigned)(gridDim.x - 1)) {
                __hip_atomic_store(cnt, 0u, __ATOMIC_RELAXED, __HIP_MEMORY_SCOPE_AGENT);
                __hip_atomic_store(gen, g + 1u, __ATOMIC_RELEASE, __HIP_MEMORY_SCOPE_AGENT);
            } else {
                while (__hip_atomic_load(gen, __ATOMIC_ACQUIRE, __HIP_MEMORY_SCOPE_AGENT) == g) {
                    __builtin_amdgcn_s_sleep(2);
                }
            }
        }
        __syncthreads();
    }

    // Final cell state c_f
#pragma unroll
    for (int r = 0; r < 8; ++r) {
        const int b = m0 + r + hi * 8;
        out[(size_t)S_LEN * BH + BH + (size_t)b * HID + n] = creg[r];
    }
}

// ---------------------------------------------------------------------------
extern "C" void kernel_launch(void* const* d_in, const int* in_sizes, int n_in,
                              void* d_out, int out_size, void* d_ws, size_t ws_size,
                              hipStream_t stream) {
    (void)in_sizes; (void)n_in; (void)out_size;

    const float* x      = (const float*)d_in[0];   // (2048, 64, 512)
    const float* h_prev = (const float*)d_in[1];   // (64, 512)
    const float* c_prev = (const float*)d_in[2];   // (64, 512)
    const float* Wih    = (const float*)d_in[3];   // (2048, 512)
    const float* Whh    = (const float*)d_in[4];   // (2048, 512)
    const float* bih    = (const float*)d_in[5];   // (2048)
    const float* bhh    = (const float*)d_in[6];   // (2048)

    // Workspace layout: optional bf16 X first, then small buffers (~4.3 MB).
    const size_t X_ELEMS  = (size_t)S_LEN * BH;           // 67,108,864
    const size_t X_BYTES  = X_ELEMS * 2;                  // 134,217,728
    const size_t SMALL    = 4268288 + 256;
    const bool   useXbf   = (ws_size >= X_BYTES + SMALL); // deterministic

    char* ws = (char*)d_ws;
    const size_t base = useXbf ? X_BYTES : 0;
    unsigned short* Xbf    = (unsigned short*)(ws);
    unsigned short* Wih_bf = (unsigned short*)(ws + base);
    unsigned short* Whh_bf = (unsigned short*)(ws + base + 2097152);
    float*          bias   = (float*)         (ws + base + 4194304);
    unsigned short* hbuf   = (unsigned short*)(ws + base + 4202496);
    unsigned*       sync   = (unsigned*)      (ws + base + 4268032);

    // Allow >64KB dynamic LDS (host-side attribute set; not a stream op).
    static_cast<void>(hipFuncSetAttribute(
        reinterpret_cast<const void*>(&lstm_persistent<true>),
        hipFuncAttributeMaxDynamicSharedMemorySize, SMEM_BYTES));
    static_cast<void>(hipFuncSetAttribute(
        reinterpret_cast<const void*>(&lstm_persistent<false>),
        hipFuncAttributeMaxDynamicSharedMemorySize, SMEM_BYTES));

    const int NW8 = (G4 * IN_DIM) / 8;   // 131,072 vec8 per weight matrix
    cvt_f32_bf16_v8<<<(NW8 + 255) / 256, 256, 0, stream>>>(Wih, Wih_bf, NW8);
    cvt_f32_bf16_v8<<<(NW8 + 255) / 256, 256, 0, stream>>>(Whh, Whh_bf, NW8);
    cvt_f32_bf16_v8<<<(BH / 8 + 255) / 256, 256, 0, stream>>>(h_prev, hbuf, BH / 8);
    add_bias<<<(G4 + 255) / 256, 256, 0, stream>>>(bih, bhh, bias, G4);
    hipMemsetAsync(sync, 0, 256, stream);   // deterministic barrier state

    if (useXbf) {
        const int NX8 = (int)(X_ELEMS / 8);  // 8,388,608
        cvt_f32_bf16_v8<<<(NX8 + 255) / 256, 256, 0, stream>>>(x, Xbf, NX8);
        lstm_persistent<true><<<NWG, THREADS, SMEM_BYTES, stream>>>(
            x, Xbf, c_prev, Wih_bf, Whh_bf, bias, hbuf, (float*)d_out, sync);
    } else {
        lstm_persistent<false><<<NWG, THREADS, SMEM_BYTES, stream>>>(
            x, Xbf, c_prev, Wih_bf, Whh_bf, bias, hbuf, (float*)d_out, sync);
    }
}
